// DotProductAttention_60945585930630
// MI455X (gfx1250) — matrix operand order repaired and checked by reading, not compiled, not run on hardware
//
#include <hip/hip_runtime.h>

// Problem constants (match reference)
constexpr int NB  = 16;
constexpr int NLQ = 2048;
constexpr int NLK = 2048;
constexpr int ND  = 256;
constexpr int NDV = 256;

typedef __attribute__((ext_vector_type(16))) __bf16 v16bf;
typedef __attribute__((ext_vector_type(8)))  float  v8f;

struct __attribute__((aligned(16))) U128 { unsigned int a, b, c, d; };

union AFrag {
  v16bf v;
  unsigned short u[16];
};

static __device__ __forceinline__ unsigned short f2bf(float f) {
  union { float f; unsigned int u; } x;
  x.f = f;
  unsigned int u = x.u;
  // round-to-nearest-even f32 -> bf16
  return (unsigned short)((u + 0x7FFFu + ((u >> 16) & 1u)) >> 16);
}

// LDS strides (in bf16 elements), padded for conflict-free b128 lane reads
constexpr int KSTR = ND + 8;   // 264 -> 528B row stride
constexpr int VSTR = 32 + 8;   // 40  -> 80B row stride
constexpr int PSTR = 40;       // per-wave P scratch row stride

__global__ __launch_bounds__(256)
void fa_wmma_kernel(const float* __restrict__ Qg,
                    const float* __restrict__ Kg,
                    const float* __restrict__ Vg,
                    const int*   __restrict__ vlen,
                    float*       __restrict__ Og,
                    int vstride) {
  __shared__ __attribute__((aligned(16))) unsigned short kbuf[32 * KSTR];       // 16.5 KB
  __shared__ __attribute__((aligned(16))) unsigned short vbuf[NDV * VSTR];      // 20 KB (transposed V)
  __shared__ __attribute__((aligned(16))) unsigned short pbuf[8 * 16 * PSTR];   // 10 KB (per-wave P)

  const int b    = blockIdx.y;
  const int lane = threadIdx.x & 31;
  const int wv   = threadIdx.x >> 5;          // wave id 0..7
  const int half = lane >> 4;                 // 0 / 1
  const int nn   = lane & 15;
  const int q0   = blockIdx.x * 128 + wv * 16;

  const int vl   = vlen[(size_t)b * vstride];
  const int kend = (vl + 31) >> 5;            // #32-key tiles (vl>=1 -> >=1)

  const float* __restrict__ Qb = Qg + ((size_t)b * NLQ) * ND;
  const float* __restrict__ Kb = Kg + ((size_t)b * NLK) * ND;
  const float* __restrict__ Vb = Vg + ((size_t)b * NLK) * NDV;

  // ---- Load this wave's 16xD Q tile into 8 bf16 A-fragments (16x32 each) ----
  // Fold the 1/sqrt(D) = 1/16 softmax scale into Q here.
  AFrag qa[8];
  {
    const float sc = 0.0625f;
    const float* qrow = Qb + (size_t)(q0 + nn) * ND;
    #pragma unroll
    for (int d = 0; d < 8; ++d) {
      const int c0 = d * 32 + 16 * half;            // K run: [c0, c0+16)
      float4 f0 = *(const float4*)(qrow + c0);
      float4 f1 = *(const float4*)(qrow + c0 + 4);
      float4 f2 = *(const float4*)(qrow + c0 + 8);
      float4 f3 = *(const float4*)(qrow + c0 + 12);
      qa[d].u[0]  = f2bf(f0.x * sc); qa[d].u[1]  = f2bf(f0.y * sc);
      qa[d].u[2]  = f2bf(f0.z * sc); qa[d].u[3]  = f2bf(f0.w * sc);
      qa[d].u[4]  = f2bf(f1.x * sc); qa[d].u[5]  = f2bf(f1.y * sc);
      qa[d].u[6]  = f2bf(f1.z * sc); qa[d].u[7]  = f2bf(f1.w * sc);
      qa[d].u[8]  = f2bf(f2.x * sc); qa[d].u[9]  = f2bf(f2.y * sc);
      qa[d].u[10] = f2bf(f2.z * sc); qa[d].u[11] = f2bf(f2.w * sc);
      qa[d].u[12] = f2bf(f3.x * sc); qa[d].u[13] = f2bf(f3.y * sc);
      qa[d].u[14] = f2bf(f3.z * sc); qa[d].u[15] = f2bf(f3.w * sc);
    }
  }

  // ---- Flash-attention state ----
  v8f zero = {};
  v8f o[16];
  #pragma unroll
  for (int g = 0; g < 16; ++g) o[g] = zero;
  float m[8], lsum[8];
  #pragma unroll
  for (int r = 0; r < 8; ++r) { m[r] = -3.0e38f; lsum[r] = 0.0f; }

  const int wbase = wv * 16 * PSTR;

  // ---- Register-pipelined K/V tile loads (tile kt+1 in flight during compute of kt) ----
  float4 rk[8], rv[8];
  {
    #pragma unroll
    for (int i = 0; i < 8; ++i) {
      const int idx = threadIdx.x + i * 256;   // 0..2047
      const int kk  = idx >> 6;
      const int cg  = (idx & 63) << 2;
      rk[i] = *(const float4*)(Kb + (size_t)kk * ND + cg);
      rv[i] = *(const float4*)(Vb + (size_t)kk * NDV + cg);
    }
  }

  for (int kt = 0; kt < kend; ++kt) {
    const int k0 = kt * 32;

    __syncthreads();   // all waves done computing on previous tile

    // ---- Convert in-flight registers -> bf16 LDS tile (K row-major, V transposed) ----
    #pragma unroll
    for (int i = 0; i < 8; ++i) {
      const int idx = threadIdx.x + i * 256;
      const int kk  = idx >> 6;
      const int cg  = (idx & 63) << 2;
      unsigned short* kd = &kbuf[kk * KSTR + cg];
      kd[0] = f2bf(rk[i].x); kd[1] = f2bf(rk[i].y);
      kd[2] = f2bf(rk[i].z); kd[3] = f2bf(rk[i].w);
      vbuf[(cg + 0) * VSTR + kk] = f2bf(rv[i].x);
      vbuf[(cg + 1) * VSTR + kk] = f2bf(rv[i].y);
      vbuf[(cg + 2) * VSTR + kk] = f2bf(rv[i].z);
      vbuf[(cg + 3) * VSTR + kk] = f2bf(rv[i].w);
    }
    // ---- Issue next tile's global loads; they complete during this tile's WMMAs ----
    if (kt + 1 < kend) {
      const int nk0 = k0 + 32;
      #pragma unroll
      for (int i = 0; i < 8; ++i) {
        const int idx = threadIdx.x + i * 256;
        const int kk  = idx >> 6;
        const int cg  = (idx & 63) << 2;
        rk[i] = *(const float4*)(Kb + (size_t)(nk0 + kk) * ND + cg);
        rv[i] = *(const float4*)(Vb + (size_t)(nk0 + kk) * NDV + cg);
      }
      // warm L2 for the tile after that
      if (kt + 2 < kend) {
        __builtin_prefetch(Kb + (size_t)(nk0 + 32) * ND + threadIdx.x * 32, 0, 0);
        __builtin_prefetch(Vb + (size_t)(nk0 + 32) * NDV + threadIdx.x * 32, 0, 0);
      }
    }
    __syncthreads();   // LDS tile visible to all waves

    // ---- S = Q*K^T : two 16x16 sub-tiles over 32 keys; fragments preloaded in
    //      chunks of 4, c0/c1 WMMAs interleaved to break the accumulator chain ----
    v8f c0 = zero, c1 = zero;
    #pragma unroll
    for (int dc = 0; dc < 8; dc += 4) {
      AFrag bk0[4], bk1[4];
      #pragma unroll
      for (int j = 0; j < 4; ++j) {
        const unsigned short* s0p = &kbuf[nn * KSTR + (dc + j) * 32 + 16 * half];
        *(U128*)&bk0[j].u[0] = *(const U128*)(s0p);
        *(U128*)&bk0[j].u[8] = *(const U128*)(s0p + 8);
        const unsigned short* s1p = &kbuf[(16 + nn) * KSTR + (dc + j) * 32 + 16 * half];
        *(U128*)&bk1[j].u[0] = *(const U128*)(s1p);
        *(U128*)&bk1[j].u[8] = *(const U128*)(s1p + 8);
      }
      #pragma unroll
      for (int j = 0; j < 4; ++j) {
        c0 = __builtin_amdgcn_wmma_f32_16x16x32_bf16(false, qa[dc + j].v, false, bk0[j].v,
                                                     (short)0, c0, false, false);
        c1 = __builtin_amdgcn_wmma_f32_16x16x32_bf16(false, qa[dc + j].v, false, bk1[j].v,
                                                     (short)0, c1, false, false);
      }
    }

    // ---- Masked online softmax (rows M = r + 8*half live across 16-lane halves) ----
    const bool msk0 = (k0 + nn)      >= vl;
    const bool msk1 = (k0 + 16 + nn) >= vl;
    float p0[8], p1[8], cf[8];
    #pragma unroll
    for (int r = 0; r < 8; ++r) {
      float s0 = msk0 ? -3.0e38f : c0[r];
      float s1 = msk1 ? -3.0e38f : c1[r];
      float t = fmaxf(s0, s1);
      t = fmaxf(t, __shfl_xor(t, 1, 32));
      t = fmaxf(t, __shfl_xor(t, 2, 32));
      t = fmaxf(t, __shfl_xor(t, 4, 32));
      t = fmaxf(t, __shfl_xor(t, 8, 32));
      const float mn = fmaxf(m[r], t);
      cf[r] = __expf(m[r] - mn);
      m[r]  = mn;
      p0[r] = __expf(s0 - mn);
      p1[r] = __expf(s1 - mn);
      float rs = p0[r] + p1[r];
      rs += __shfl_xor(rs, 1, 32);
      rs += __shfl_xor(rs, 2, 32);
      rs += __shfl_xor(rs, 4, 32);
      rs += __shfl_xor(rs, 8, 32);
      lsum[r] = lsum[r] * cf[r] + rs;
    }
    #pragma unroll
    for (int g = 0; g < 16; ++g) {
      #pragma unroll
      for (int r = 0; r < 8; ++r) o[g][r] = o[g][r] * cf[r];
    }

    // ---- C-layout P -> A-layout via per-wave LDS scratch ----
    #pragma unroll
    for (int r = 0; r < 8; ++r) {
      const int Mrow = r + 8 * half;
      pbuf[wbase + Mrow * PSTR + nn]      = f2bf(p0[r]);
      pbuf[wbase + Mrow * PSTR + 16 + nn] = f2bf(p1[r]);
    }
    asm volatile("s_wait_dscnt 0" ::: "memory");
    AFrag pa;
    {
      const unsigned short* src = &pbuf[wbase + nn * PSTR];
      *(U128*)&pa.u[0] = *(const U128*)(src + 16 * half);       // K = 16h .. 16h+7
      *(U128*)&pa.u[8] = *(const U128*)(src + 16 * half + 8);   // K = 16h+8 .. 16h+15
    }

    // ---- O += P(16x32) * V(32x16), 16 column groups, fragments chunked by 4 ----
    #pragma unroll
    for (int gc = 0; gc < 16; gc += 4) {
      AFrag bv[4];
      #pragma unroll
      for (int j = 0; j < 4; ++j) {
        const unsigned short* sp = &vbuf[(16 * (gc + j) + nn) * VSTR + 16 * half];
        *(U128*)&bv[j].u[0] = *(const U128*)(sp);
        *(U128*)&bv[j].u[8] = *(const U128*)(sp + 8);
      }
      #pragma unroll
      for (int j = 0; j < 4; ++j)
        o[gc + j] = __builtin_amdgcn_wmma_f32_16x16x32_bf16(false, pa.v, false, bv[j].v,
                                                            (short)0, o[gc + j], false, false);
    }
  }

  // ---- Normalize and store fp32 output ----
  float inv[8];
  #pragma unroll
  for (int r = 0; r < 8; ++r) inv[r] = 1.0f / lsum[r];
  #pragma unroll
  for (int g = 0; g < 16; ++g) {
    #pragma unroll
    for (int r = 0; r < 8; ++r) {
      const size_t row = (size_t)b * NLQ + (size_t)(q0 + r + 8 * half);
      Og[row * NDV + 16 * g + nn] = o[g][r] * inv[r];
    }
  }
}

extern "C" void kernel_launch(void* const* d_in, const int* in_sizes, int n_in,
                              void* d_out, int out_size, void* d_ws, size_t ws_size,
                              hipStream_t stream) {
  const float* Q  = (const float*)d_in[0];
  const float* K  = (const float*)d_in[1];
  const float* V  = (const float*)d_in[2];
  const int*   vl = (const int*)d_in[3];
  // Hedge: if harness exposes int64 valid_len as 2*B int32 elements, step by 2
  // (low word of each little-endian int64 is correct for values <= 2048).
  const int vstride = (n_in > 3 && in_sizes[3] >= 2 * NB) ? 2 : 1;
  float* out = (float*)d_out;

  dim3 grid(NLQ / 128, NB);
  dim3 block(256);
  fa_wmma_kernel<<<grid, block, 0, stream>>>(Q, K, V, vl, out, vstride);
}
